// TransformerBlock_86698209837641
// MI455X (gfx1250) — matrix operand order repaired and checked by reading, not compiled, not run on hardware
//
#include <hip/hip_runtime.h>

#define D_MODEL 2048
#define D_FF    8192
#define SEQ     2048
#define BATCH   2
#define NHEADS  16
#define HDIM    128
#define MROWS   (BATCH*SEQ)   // 4096 token rows
#define LN_EPS  1e-5f

typedef __attribute__((ext_vector_type(16))) __bf16 bf16x16;
typedef __attribute__((ext_vector_type(8)))  float  f32x8;
typedef int v4i __attribute__((__vector_size__(16)));
typedef v4i __attribute__((address_space(1)))* gbl_v4i_p;
typedef v4i __attribute__((address_space(3)))* lds_v4i_p;

union Frag { bf16x16 v; unsigned int u[8]; };

// gfx1250 async global->LDS path (ASYNCcnt), guarded so the file compiles
// on toolchains that don't declare the builtins.
#if defined(__has_builtin)
#if __has_builtin(__builtin_amdgcn_global_load_async_to_lds_b128) && \
    __has_builtin(__builtin_amdgcn_s_wait_asynccnt)
#define USE_ASYNC_COPY 1
#endif
#endif
#ifndef USE_ASYNC_COPY
#define USE_ASYNC_COPY 0
#endif

__device__ __forceinline__ unsigned short f2bf(float f) {
    union { float f; unsigned int u; } c; c.f = f;
    unsigned int r = (c.u + 0x7FFFu + ((c.u >> 16) & 1u)) >> 16;
    return (unsigned short)r;
}

__device__ __forceinline__ unsigned int lds32(const unsigned short* p) {
    return *reinterpret_cast<const unsigned int*>(p);
}

// copy 32 contiguous bytes global->LDS
__device__ __forceinline__ void stage32B(const unsigned short* g, unsigned short* l) {
#if USE_ASYNC_COPY
    gbl_v4i_p gs = (gbl_v4i_p)g;
    lds_v4i_p ls = (lds_v4i_p)l;
    __builtin_amdgcn_global_load_async_to_lds_b128(gs, ls, 0, 0);
    __builtin_amdgcn_global_load_async_to_lds_b128(gs, ls, 16, 0);
#else
    const uint4* src = reinterpret_cast<const uint4*>(g);
    uint4 v0 = src[0], v1 = src[1];
    uint4* dst = reinterpret_cast<uint4*>(l);
    dst[0] = v0; dst[1] = v1;
#endif
}

__device__ __forceinline__ void stage_fence() {
#if USE_ASYNC_COPY
    __builtin_amdgcn_s_wait_asynccnt(0);
#endif
}

// ---------------------------------------------------------------------------
// fp32 -> bf16 weight conversion (one-time per launch, amortized)
// ---------------------------------------------------------------------------
__global__ void cvt_bf16_kernel(const float* __restrict__ in,
                                unsigned short* __restrict__ out, size_t n) {
    size_t i = (size_t)blockIdx.x * blockDim.x + threadIdx.x;
    size_t stride = (size_t)gridDim.x * blockDim.x;
    for (; i < n; i += stride) out[i] = f2bf(in[i]);
}

// ---------------------------------------------------------------------------
// LayerNorm (optionally fused residual add), writes bf16 activation
// and optionally the fp32 residual stream
// ---------------------------------------------------------------------------
template<bool ADD, bool WRITE_X1>
__global__ void __launch_bounds__(256) ln_kernel(
        const float* __restrict__ x,
        const float* __restrict__ addp,
        const float* __restrict__ g,
        const float* __restrict__ bcoef,
        unsigned short* __restrict__ y,
        float* __restrict__ x1) {
    __shared__ float s1[256];
    __shared__ float s2[256];
    const int row = blockIdx.x;
    const int tid = threadIdx.x;
    const size_t base = (size_t)row * D_MODEL;
    float loc[8];
    float sum = 0.f, sq = 0.f;
#pragma unroll
    for (int i = 0; i < 8; ++i) {
        int col = tid + i * 256;
        float v = x[base + col];
        if (ADD) v += addp[base + col];
        loc[i] = v; sum += v; sq += v * v;
    }
    s1[tid] = sum; s2[tid] = sq;
    __syncthreads();
    for (int off = 128; off > 0; off >>= 1) {
        if (tid < off) { s1[tid] += s1[tid + off]; s2[tid] += s2[tid + off]; }
        __syncthreads();
    }
    float mean = s1[0] * (1.f / D_MODEL);
    float var  = s2[0] * (1.f / D_MODEL) - mean * mean;
    float rstd = rsqrtf(var + LN_EPS);
#pragma unroll
    for (int i = 0; i < 8; ++i) {
        int col = tid + i * 256;
        float ln = (loc[i] - mean) * rstd * g[col] + bcoef[col];
        y[base + col] = f2bf(ln);
        if (WRITE_X1) x1[base + col] = loc[i];
    }
}

// ---------------------------------------------------------------------------
// bf16 WMMA GEMM: C[M,N] = A[M,K] * B[K,N] + bias (+ReLU) (+residual)
// 128x128 workgroup tile, 8 waves (2x4), each wave 64x32, K-step 32.
// ---------------------------------------------------------------------------
template<bool RELU, bool RESIDUAL, bool OUT_BF16>
__global__ void __launch_bounds__(256, 1) gemm_wmma_kernel(
        const unsigned short* __restrict__ A,
        const unsigned short* __restrict__ Bw,
        const float* __restrict__ bias,
        const float* __restrict__ res,
        void* __restrict__ Cout,
        int M, int N, int K) {
    constexpr int LSTR = 40;  // padded row stride (elements): 80B, 16B-aligned
    __shared__ __align__(16) unsigned short aT[128 * LSTR]; // [row][k]
    __shared__ __align__(16) unsigned short bT[128 * LSTR]; // [n][k] (transposed)

    const int tid  = threadIdx.x;
    const int wave = tid >> 5, lane = tid & 31;
    const int hf   = lane >> 4, lrow = lane & 15;
    const int wm   = (wave & 1) * 64;
    const int wn   = (wave >> 1) * 32;
    const int m0   = blockIdx.y * 128;
    const int n0   = blockIdx.x * 128;

    f32x8 acc[4][2];
#pragma unroll
    for (int i = 0; i < 4; ++i) { acc[i][0] = {}; acc[i][1] = {}; }

    const int ar   = tid >> 1;            // A stage: row 0..127
    const int aseg = (tid & 1) * 16;      // k segment 0/16
    const int bk   = tid & 31;            // B stage: k 0..31
    const int bns  = (tid >> 5) * 16;     // n segment 0..112

    for (int k0 = 0; k0 < K; k0 += 32) {
        if (k0 + 32 < K) { // prefetch next K tiles -> global_prefetch_b8
            __builtin_prefetch(A  + (size_t)(m0 + ar) * K + k0 + 32 + aseg, 0, 1);
            __builtin_prefetch(Bw + (size_t)(k0 + 32 + bk) * N + n0 + bns, 0, 1);
        }
        // stage A (row-major, contiguous dest -> async global->LDS path)
        stage32B(A + (size_t)(m0 + ar) * K + k0 + aseg, &aT[ar * LSTR + aseg]);
        { // stage B transposed: bT[n][k] = B[k0+k][n0+n] (16-bit scatter)
            const uint4* src = reinterpret_cast<const uint4*>(
                Bw + (size_t)(k0 + bk) * N + n0 + bns);
            uint4 v0 = src[0], v1 = src[1];
            const unsigned short* pv = reinterpret_cast<const unsigned short*>(&v0);
#pragma unroll
            for (int i = 0; i < 8; ++i) bT[(bns + i) * LSTR + bk] = pv[i];
            pv = reinterpret_cast<const unsigned short*>(&v1);
#pragma unroll
            for (int i = 0; i < 8; ++i) bT[(bns + 8 + i) * LSTR + bk] = pv[i];
        }
        stage_fence();
        __syncthreads();

        Frag bfr[2];
#pragma unroll
        for (int ni = 0; ni < 2; ++ni) {
            const unsigned short* bp = &bT[(wn + ni * 16 + lrow) * LSTR + hf * 16];
#pragma unroll
            for (int p = 0; p < 8; ++p) bfr[ni].u[p] = lds32(bp + 2 * p);
        }
#pragma unroll
        for (int mi = 0; mi < 4; ++mi) {
            Frag afr;
            const unsigned short* ap = &aT[(wm + mi * 16 + lrow) * LSTR];
#pragma unroll
            for (int p = 0; p < 4; ++p) {
                afr.u[p]     = lds32(ap + hf * 16 + 2 * p);
                afr.u[4 + p] = lds32(ap + hf * 16 + 8 + 2 * p);
            }
            acc[mi][0] = __builtin_amdgcn_wmma_f32_16x16x32_bf16(
                false, afr.v, false, bfr[0].v, (short)0, acc[mi][0], false, false);
            acc[mi][1] = __builtin_amdgcn_wmma_f32_16x16x32_bf16(
                false, afr.v, false, bfr[1].v, (short)0, acc[mi][1], false, false);
        }
        __syncthreads();
    }

#pragma unroll
    for (int mi = 0; mi < 4; ++mi)
#pragma unroll
        for (int ni = 0; ni < 2; ++ni)
#pragma unroll
            for (int r = 0; r < 8; ++r) {
                int gm = m0 + wm + mi * 16 + r + hf * 8;
                int gn = n0 + wn + ni * 16 + lrow;
                float vv = acc[mi][ni][r] + bias[gn];
                if (RELU) vv = fmaxf(vv, 0.f);
                if (RESIDUAL) vv += res[(size_t)gm * N + gn];
                if (OUT_BF16)
                    reinterpret_cast<unsigned short*>(Cout)[(size_t)gm * N + gn] = f2bf(vv);
                else
                    reinterpret_cast<float*>(Cout)[(size_t)gm * N + gn] = vv;
            }
}

// ---------------------------------------------------------------------------
// Flash attention: one block = (128 query rows, head, batch); 8 waves,
// each wave owns 16 query rows. QK^T and P*V via WMMA, online softmax.
// ---------------------------------------------------------------------------
__global__ void __launch_bounds__(256, 1) attn_flash_kernel(
        const unsigned short* __restrict__ q,
        const unsigned short* __restrict__ k,
        const unsigned short* __restrict__ v,
        float* __restrict__ o_out) {
    constexpr int KSTR = 136;  // 272B rows, 16B aligned
    constexpr int VSTR = 40;
    constexpr int PSTR = 36;
    __shared__ __align__(16) unsigned short kT[32 * KSTR];   // [t][hd]
    __shared__ __align__(16) unsigned short vT[128 * VSTR];  // [hd][t] transposed
    __shared__ __align__(16) unsigned short pT[8 * 16 * PSTR];

    const int tid  = threadIdx.x;
    const int wave = tid >> 5, lane = tid & 31;
    const int hf   = lane >> 4, lrow = lane & 15;
    const int head = blockIdx.y, b = blockIdx.z;
    const int srow0 = blockIdx.x * 128 + wave * 16;
    const size_t qrowbase =
        ((size_t)b * SEQ + srow0 + lrow) * D_MODEL + head * HDIM;

    // Q A-fragments held in registers for the whole kernel (4 x 16x32 slices)
    Frag qf[4];
#pragma unroll
    for (int f = 0; f < 4; ++f) {
        const unsigned short* qp = q + qrowbase + f * 32 + hf * 16;
#pragma unroll
        for (int p = 0; p < 4; ++p) {
            qf[f].u[p]     = *reinterpret_cast<const unsigned int*>(qp + 2 * p);
            qf[f].u[4 + p] = *reinterpret_cast<const unsigned int*>(qp + 8 + 2 * p);
        }
    }

    f32x8 oacc[8];
#pragma unroll
    for (int i = 0; i < 8; ++i) oacc[i] = {};
    float mrun[8], lrun[8];
#pragma unroll
    for (int r = 0; r < 8; ++r) { mrun[r] = -1e30f; lrun[r] = 0.f; }

    const int st   = tid >> 3;            // K stage: t 0..31
    const int sseg = (tid & 7) * 16;      // hd segment
    const int vt   = tid & 31;            // V stage: t 0..31
    const int vseg = (tid >> 5) * 16;     // hd segment

    unsigned short* pW = &pT[wave * 16 * PSTR];
    const float SCALE = 0.08838834764831845f; // 1/sqrt(128)

    for (int t0 = 0; t0 < SEQ; t0 += 32) {
        // stage K tile row-major (contiguous dest -> async path)
        stage32B(k + ((size_t)b * SEQ + t0 + st) * D_MODEL + head * HDIM + sseg,
                 &kT[st * KSTR + sseg]);
        { // stage V transposed: vT[hd][t] (16-bit scatter)
            const uint4* src = reinterpret_cast<const uint4*>(
                v + ((size_t)b * SEQ + t0 + vt) * D_MODEL + head * HDIM + vseg);
            uint4 v0 = src[0], v1 = src[1];
            const unsigned short* pv = reinterpret_cast<const unsigned short*>(&v0);
#pragma unroll
            for (int i = 0; i < 8; ++i) vT[(vseg + i) * VSTR + vt] = pv[i];
            pv = reinterpret_cast<const unsigned short*>(&v1);
#pragma unroll
            for (int i = 0; i < 8; ++i) vT[(vseg + 8 + i) * VSTR + vt] = pv[i];
        }
        stage_fence();
        __syncthreads();

        // scores for 32 key columns: two 16x16 tiles, K-dim = 128 (4 wmma each)
        f32x8 c0 = {}, c1 = {};
#pragma unroll
        for (int f = 0; f < 4; ++f) {
            Frag b0, b1;
            const unsigned short* kp0 = &kT[lrow * KSTR + f * 32 + hf * 16];
            const unsigned short* kp1 = &kT[(16 + lrow) * KSTR + f * 32 + hf * 16];
#pragma unroll
            for (int p = 0; p < 8; ++p) {
                b0.u[p] = lds32(kp0 + 2 * p);
                b1.u[p] = lds32(kp1 + 2 * p);
            }
            c0 = __builtin_amdgcn_wmma_f32_16x16x32_bf16(
                false, qf[f].v, false, b0.v, (short)0, c0, false, false);
            c1 = __builtin_amdgcn_wmma_f32_16x16x32_bf16(
                false, qf[f].v, false, b1.v, (short)0, c1, false, false);
        }

        // online softmax over the 32 columns (row = r + hf*8, cols across 16 lanes)
#pragma unroll
        for (int r = 0; r < 8; ++r) {
            float s0 = c0[r] * SCALE, s1 = c1[r] * SCALE;
            float mt = fmaxf(s0, s1);
#pragma unroll
            for (int msk = 1; msk < 16; msk <<= 1)
                mt = fmaxf(mt, __shfl_xor(mt, msk, 32));
            float mnew  = fmaxf(mrun[r], mt);
            float p0    = __expf(s0 - mnew);
            float p1    = __expf(s1 - mnew);
            float alpha = __expf(mrun[r] - mnew);
            float rs = p0 + p1;
#pragma unroll
            for (int msk = 1; msk < 16; msk <<= 1)
                rs += __shfl_xor(rs, msk, 32);
            lrun[r] = lrun[r] * alpha + rs;
            mrun[r] = mnew;
#pragma unroll
            for (int ht = 0; ht < 8; ++ht) oacc[ht][r] *= alpha;
            int prow = r + hf * 8;
            pW[prow * PSTR + lrow]      = f2bf(p0);
            pW[prow * PSTR + 16 + lrow] = f2bf(p1);
        }

        // rebuild P as an A-fragment (C layout -> A layout via per-wave LDS)
        Frag pf;
        {
            const unsigned short* pp = &pW[lrow * PSTR];
#pragma unroll
            for (int p = 0; p < 4; ++p) {
                pf.u[p]     = lds32(pp + hf * 16 + 2 * p);
                pf.u[4 + p] = lds32(pp + hf * 16 + 8 + 2 * p);
            }
        }
        // o += P * V  (8 output column tiles across head dim 128)
#pragma unroll
        for (int ht = 0; ht < 8; ++ht) {
            Frag bv;
            const unsigned short* vp = &vT[(ht * 16 + lrow) * VSTR + hf * 16];
#pragma unroll
            for (int p = 0; p < 8; ++p) bv.u[p] = lds32(vp + 2 * p);
            oacc[ht] = __builtin_amdgcn_wmma_f32_16x16x32_bf16(
                false, pf.v, false, bv.v, (short)0, oacc[ht], false, false);
        }
        __syncthreads();
    }

#pragma unroll
    for (int ht = 0; ht < 8; ++ht)
#pragma unroll
        for (int r = 0; r < 8; ++r) {
            int srow = srow0 + r + hf * 8;
            size_t idx = ((size_t)b * SEQ + srow) * D_MODEL + head * HDIM +
                         ht * 16 + lrow;
            o_out[idx] = oacc[ht][r] / lrun[r];
        }
}

// ---------------------------------------------------------------------------
extern "C" void kernel_launch(void* const* d_in, const int* in_sizes, int n_in,
                              void* d_out, int out_size, void* d_ws, size_t ws_size,
                              hipStream_t stream) {
    (void)in_sizes; (void)n_in; (void)out_size; (void)ws_size;
    const float* x   = (const float*)d_in[0];
    const float* Wq  = (const float*)d_in[1];
    const float* bq  = (const float*)d_in[2];
    const float* Wk  = (const float*)d_in[3];
    const float* bk  = (const float*)d_in[4];
    const float* Wv  = (const float*)d_in[5];
    const float* bv  = (const float*)d_in[6];
    const float* W1  = (const float*)d_in[7];
    const float* b1  = (const float*)d_in[8];
    const float* W2  = (const float*)d_in[9];
    const float* b2  = (const float*)d_in[10];
    const float* g1  = (const float*)d_in[11];
    const float* be1 = (const float*)d_in[12];
    const float* g2  = (const float*)d_in[13];
    const float* be2 = (const float*)d_in[14];
    float* out = (float*)d_out;

    char* ws = (char*)d_ws;
    size_t off = 0;
    auto alloc = [&](size_t bytes) -> char* {
        char* p = ws + off;
        off += (bytes + 255) & ~(size_t)255;
        return p;
    };
    unsigned short* WqB  = (unsigned short*)alloc((size_t)D_MODEL * D_MODEL * 2);
    unsigned short* WkB  = (unsigned short*)alloc((size_t)D_MODEL * D_MODEL * 2);
    unsigned short* WvB  = (unsigned short*)alloc((size_t)D_MODEL * D_MODEL * 2);
    unsigned short* W1B  = (unsigned short*)alloc((size_t)D_MODEL * D_FF * 2);
    unsigned short* W2B  = (unsigned short*)alloc((size_t)D_FF * D_MODEL * 2);
    unsigned short* h1B  = (unsigned short*)alloc((size_t)MROWS * D_MODEL * 2);
    unsigned short* qB   = (unsigned short*)alloc((size_t)MROWS * D_MODEL * 2);
    unsigned short* kB   = (unsigned short*)alloc((size_t)MROWS * D_MODEL * 2);
    unsigned short* vB   = (unsigned short*)alloc((size_t)MROWS * D_MODEL * 2);
    float*          attnF= (float*)alloc((size_t)MROWS * D_MODEL * 4);
    float*          x1F  = (float*)alloc((size_t)MROWS * D_MODEL * 4);
    unsigned short* h2B  = (unsigned short*)alloc((size_t)MROWS * D_MODEL * 2);
    unsigned short* ffnB = (unsigned short*)alloc((size_t)MROWS * D_FF * 2);

    // 1. weights -> bf16
    cvt_bf16_kernel<<<2048, 256, 0, stream>>>(Wq, WqB, (size_t)D_MODEL * D_MODEL);
    cvt_bf16_kernel<<<2048, 256, 0, stream>>>(Wk, WkB, (size_t)D_MODEL * D_MODEL);
    cvt_bf16_kernel<<<2048, 256, 0, stream>>>(Wv, WvB, (size_t)D_MODEL * D_MODEL);
    cvt_bf16_kernel<<<4096, 256, 0, stream>>>(W1, W1B, (size_t)D_MODEL * D_FF);
    cvt_bf16_kernel<<<4096, 256, 0, stream>>>(W2, W2B, (size_t)D_FF * D_MODEL);

    // 2. LN1 -> bf16 activations
    ln_kernel<false, false><<<MROWS, 256, 0, stream>>>(x, nullptr, g1, be1, h1B, nullptr);

    // 3. QKV projections (WMMA)
    dim3 gqkv(D_MODEL / 128, MROWS / 128);
    gemm_wmma_kernel<false, false, true><<<gqkv, 256, 0, stream>>>(
        h1B, WqB, bq, nullptr, qB, MROWS, D_MODEL, D_MODEL);
    gemm_wmma_kernel<false, false, true><<<gqkv, 256, 0, stream>>>(
        h1B, WkB, bk, nullptr, kB, MROWS, D_MODEL, D_MODEL);
    gemm_wmma_kernel<false, false, true><<<gqkv, 256, 0, stream>>>(
        h1B, WvB, bv, nullptr, vB, MROWS, D_MODEL, D_MODEL);

    // 4. flash attention (WMMA)
    attn_flash_kernel<<<dim3(SEQ / 128, NHEADS, BATCH), 256, 0, stream>>>(
        qB, kB, vB, attnF);

    // 5. residual add + LN2 (writes x1 fp32 and h2 bf16)
    ln_kernel<true, true><<<MROWS, 256, 0, stream>>>(x, attnF, g2, be2, h2B, x1F);

    // 6. FFN
    gemm_wmma_kernel<true, false, true><<<dim3(D_FF / 128, MROWS / 128), 256, 0, stream>>>(
        h2B, W1B, b1, nullptr, ffnB, MROWS, D_FF, D_MODEL);
    gemm_wmma_kernel<false, true, false><<<dim3(D_MODEL / 128, MROWS / 128), 256, 0, stream>>>(
        ffnB, W2B, b2, x1F, out, MROWS, D_MODEL, D_FF);
}
